// SparseMultiHeadAttention_10514079940986
// MI455X (gfx1250) — matrix operand order repaired and checked by reading, not compiled, not run on hardware
//
#include <hip/hip_runtime.h>

// ---------------------------------------------------------------------------
// Sparse (banded) multi-head attention for MI455X / gfx1250, wave32 + WMMA.
// S=4096, D_MODEL=1024, H=16, DK=64, WINDOW=128, GLOBAL=0.
//
// Pipeline:
//   0) pre-convert q/k/v and wq/wk/wv/wo to f16 (memory-bound problem:
//      halve bytes, remove per-wave cvt from GEMM hot loops)
//   1) QKV projection: wave = 32x64 tile, v_wmma_f32_16x16x32_f16,
//      Q/K row-major f16 [H][S][64], V transposed f16 [H][64][S]
//   2) RoPE on Q (x softmax scale) and K
//   3) banded attention: wave per (head, 16-query block), 16x288 LDS tile
//   4) output projection (f32 out)
// ---------------------------------------------------------------------------

#define S_LEN   4096
#define DMODEL  1024
#define NHEADS  16
#define DHEAD   64
#define WIN     128

typedef __attribute__((ext_vector_type(16))) _Float16 v16h;
typedef __attribute__((ext_vector_type(8)))  _Float16 v8h;
typedef __attribute__((ext_vector_type(8)))  float    v8f;
typedef __attribute__((ext_vector_type(4)))  float    v4f;

static __device__ __forceinline__ v16h cat8(v8h lo, v8h hi) {
  return __builtin_shufflevector(lo, hi, 0,1,2,3,4,5,6,7,8,9,10,11,12,13,14,15);
}

static __device__ __forceinline__ int iclamp(int x, int lo, int hi) {
  return x < lo ? lo : (x > hi ? hi : x);
}

// ---------------------------------------------------------------------------
// Kernel 0: bulk f32 -> f16 conversion (8 elements / thread).
// ---------------------------------------------------------------------------
__global__ __launch_bounds__(256)
void cvt_f16_kernel(const float* __restrict__ src, _Float16* __restrict__ dst) {
  const size_t i = ((size_t)blockIdx.x * blockDim.x + threadIdx.x) * 8;
  v4f a = *(const v4f*)(src + i);
  v4f b = *(const v4f*)(src + i + 4);
  v8h o;
#pragma unroll
  for (int t = 0; t < 4; ++t) {
    o[t]     = (_Float16)a[t];
    o[4 + t] = (_Float16)b[t];
  }
  *(v8h*)(dst + i) = o;
}

// ---------------------------------------------------------------------------
// Kernel 1: fused QKV projection.  One wave computes a 32(row) x 64(col) tile;
// both row-fragments share the 4 B-fragments each K-step (8 WMMA / step).
//   Out[s][n] = sum_k X[s][k] * W[n][k] + b[n]
// Q,K -> row-major f16 [H][S][DHEAD]; V -> transposed f16 [H][DHEAD][S].
// ---------------------------------------------------------------------------
__global__ __launch_bounds__(32)
void qkv_proj_kernel(const _Float16* __restrict__ Xh,   // [3][S][DMODEL] f16
                     const _Float16* __restrict__ Wh,   // [4][DMODEL][DMODEL] f16
                     const float* __restrict__ bq, const float* __restrict__ bk,
                     const float* __restrict__ bv,
                     _Float16* __restrict__ Qr, _Float16* __restrict__ Kr,
                     _Float16* __restrict__ Vt) {
  const int l  = threadIdx.x;
  const int lm = l & 15;        // lane within half-wave: N column / M row select
  const int lh = l >> 4;        // half-wave group: K-base select
  const int s0  = blockIdx.x * 32;
  const int n0  = blockIdx.y * 64;
  const int mat = blockIdx.z;   // 0=Q 1=K 2=V (uniform per block)

  const _Float16* X = Xh + (size_t)mat * S_LEN  * DMODEL;
  const _Float16* W = Wh + (size_t)mat * DMODEL * DMODEL;
  const float*    B = (mat == 0) ? bq : (mat == 1) ? bk : bv;

  v8f acc[2][4] = {};

  const _Float16* xrow0 = X + (size_t)(s0 + lm) * DMODEL;
  const _Float16* xrow1 = xrow0 + (size_t)16 * DMODEL;

  for (int k0 = 0; k0 < DMODEL; k0 += 32) {
    __builtin_prefetch(xrow0 + k0 + 64, 0, 1);          // global_prefetch_b8
    // A-fragments: lane holds K = {b..b+7, b+16..b+23}, b = lh*8
    v16h a0 = cat8(*(const v8h*)(xrow0 + k0 + lh * 8),
                   *(const v8h*)(xrow0 + k0 + lh * 8 + 16));
    v16h a1 = cat8(*(const v8h*)(xrow1 + k0 + lh * 8),
                   *(const v8h*)(xrow1 + k0 + lh * 8 + 16));
#pragma unroll
    for (int nt = 0; nt < 4; ++nt) {
      // B-fragment: lane = column n, K = {b..b+7, b+16..b+23}, b = lh*8 (matches A)
      v16h b = cat8(*(const v8h*)(W + (size_t)(n0 + nt * 16 + lm) * DMODEL + k0 + lh * 8),
                    *(const v8h*)(W + (size_t)(n0 + nt * 16 + lm) * DMODEL + k0 + lh * 8 + 16));
      acc[0][nt] = __builtin_amdgcn_wmma_f32_16x16x32_f16(
          false, a0, false, b, (short)0, acc[0][nt], false, false);
      acc[1][nt] = __builtin_amdgcn_wmma_f32_16x16x32_f16(
          false, a1, false, b, (short)0, acc[1][nt], false, false);
    }
  }

  // Writeback. D layout: VGPR vv -> row m = vv + 8*lh, column = lm.
#pragma unroll
  for (int rt = 0; rt < 2; ++rt) {
    const int sb = s0 + rt * 16;
#pragma unroll
    for (int nt = 0; nt < 4; ++nt) {
      const int n  = n0 + nt * 16 + lm;
      const float bb = B[n];
      const int h = n >> 6;
      const int d = n & 63;
      if (mat < 2) {
        _Float16* dst = ((mat == 0) ? Qr : Kr);
#pragma unroll
        for (int vv = 0; vv < 8; ++vv) {
          const int m = vv + lh * 8;
          dst[((size_t)h * S_LEN + sb + m) * DHEAD + d] =
              (_Float16)(acc[rt][nt][vv] + bb);
        }
      } else {
        // V transposed: Vt[h][d][sb + lh*8 .. +7] -- one contiguous 16B store
        v8h pack;
#pragma unroll
        for (int vv = 0; vv < 8; ++vv) pack[vv] = (_Float16)(acc[rt][nt][vv] + bb);
        *(v8h*)(Vt + ((size_t)h * DHEAD + d) * S_LEN + sb + lh * 8) = pack;
      }
    }
  }
}

// ---------------------------------------------------------------------------
// Kernel 2: RoPE in place on Q and K (row-major [H][S][DHEAD], f16).
//   x_rope[d] = -x[2d+1] (d<32) ;  x[2(d-32)] (d>=32)
//   y[d] = x[d]*cos(s*f[d%32]) + x_rope[d]*sin(s*f[d%32]),  f[j]=10000^(-j/32)
// Q additionally scaled by 1/sqrt(DK) (folds the softmax scale).
// ---------------------------------------------------------------------------
__global__ __launch_bounds__(256)
void rope_kernel(_Float16* __restrict__ Qr, _Float16* __restrict__ Kr) {
  const int idx = blockIdx.x * blockDim.x + threadIdx.x;   // 0 .. NHEADS*S-1
  const bool isQ = (blockIdx.y == 0);
  _Float16* row = (isQ ? Qr : Kr) + (size_t)idx * DHEAD;
  const int s = idx & (S_LEN - 1);
  const float outscale = isQ ? 0.125f : 1.0f;              // 1/sqrt(64)

  float x[DHEAD];
#pragma unroll
  for (int d = 0; d < DHEAD; ++d) x[d] = (float)row[d];

  const float nln_over32 = -9.210340371976184f / 32.0f;    // -ln(10000)/32
#pragma unroll
  for (int d = 0; d < DHEAD; ++d) {
    const int j = d & 31;
    const float ang = (float)s * __expf((float)j * nln_over32);
    const float c = __cosf(ang), sn = __sinf(ang);
    const float xr = (d < 32) ? -x[2 * d + 1] : x[2 * (d - 32)];
    row[d] = (_Float16)((x[d] * c + xr * sn) * outscale);
  }
}

// ---------------------------------------------------------------------------
// Kernel 3: banded attention. One wave per (head, 16-query block).
// Keys live in j in [qb*16-128, qb*16+143]  (272 columns; padded to 288).
// ---------------------------------------------------------------------------
__global__ __launch_bounds__(32)
void attn_kernel(const _Float16* __restrict__ Qr, const _Float16* __restrict__ Kr,
                 const _Float16* __restrict__ Vt, _Float16* __restrict__ AO) {
  __shared__ float    sS[16][288];     // f32 scores (already scaled via Q)
  __shared__ _Float16 sP[16][288];     // f16 probabilities (unnormalized)
  __shared__ float    rsinv[16];       // per-row 1/sum

  const int l  = threadIdx.x;
  const int lm = l & 15;
  const int lh = l >> 4;
  const int qb = blockIdx.x;                 // query block (256 total)
  const int h  = blockIdx.y;                 // head
  const int jbase = qb * 16 - WIN;           // first (possibly negative) key col

  // --- Q fragments (rows = queries, K-dim = head dim, 2 chunks of 32) ---
  const _Float16* qrow = Qr + ((size_t)h * S_LEN + qb * 16 + lm) * DHEAD;
  v16h aq0 = cat8(*(const v8h*)(qrow + lh * 16),      *(const v8h*)(qrow + lh * 16 + 8));
  v16h aq1 = cat8(*(const v8h*)(qrow + 32 + lh * 16), *(const v8h*)(qrow + 32 + lh * 16 + 8));

  // --- scores: 18 key tiles of 16 (cover 288 cols), 2 WMMA each ---
  for (int jt = 0; jt < 18; ++jt) {
    const int j0  = jbase + jt * 16;
    const int j0c = iclamp(j0, 0, S_LEN - 16);          // safe load; masked later
    const _Float16* krow = Kr + ((size_t)h * S_LEN + j0c + lm) * DHEAD + lh * 16;
    v16h bk0 = *(const v16h*)(krow);
    v16h bk1 = *(const v16h*)(krow + 32);
    v8f c = {};
    c = __builtin_amdgcn_wmma_f32_16x16x32_f16(false, aq0, false, bk0, (short)0, c, false, false);
    c = __builtin_amdgcn_wmma_f32_16x16x32_f16(false, aq1, false, bk1, (short)0, c, false, false);
#pragma unroll
    for (int vv = 0; vv < 8; ++vv) sS[vv + lh * 8][jt * 16 + lm] = c[vv];
  }
  __syncthreads();

  // --- per-row softmax over the band (lanes 0-15, one row each) ---
  if (l < 16) {
    const int i = qb * 16 + l;
    float mx = -3.4e38f;
    for (int t = 0; t < 272; ++t) {
      const int j = jbase + t;
      const bool valid = (j >= 0) && (j < S_LEN) && (j >= i - WIN) && (j <= i + WIN);
      if (valid) mx = fmaxf(mx, sS[l][t]);
    }
    float sum = 0.0f;
    for (int t = 0; t < 288; ++t) {
      const int j = jbase + t;
      const bool valid = (t < 272) && (j >= 0) && (j < S_LEN) &&
                         (j >= i - WIN) && (j <= i + WIN);
      const float e = valid ? __expf(sS[l][t] - mx) : 0.0f;
      sum += e;
      sP[l][t] = (_Float16)e;
    }
    rsinv[l] = 1.0f / sum;
  }
  __syncthreads();

  // --- PV: 9 key chunks of 32, 4 output column tiles of 16 ---
  for (int dt = 0; dt < 4; ++dt) {
    v8f acc = {};
    const _Float16* vrow = Vt + ((size_t)h * DHEAD + dt * 16 + lm) * S_LEN;
    for (int jc = 0; jc < 9; ++jc) {
      // A-fragment straight from LDS (contiguous, 16B aligned)
      v16h ap = cat8(*(const v8h*)(&sP[lm][jc * 32 + lh * 16]),
                     *(const v8h*)(&sP[lm][jc * 32 + lh * 16 + 8]));
      // B-fragment from transposed V: lane = output column d, contiguous j
      int js = iclamp(jbase + jc * 32 + lh * 16, 0, S_LEN - 16);
      v16h bv = *(const v16h*)(vrow + js);
      acc = __builtin_amdgcn_wmma_f32_16x16x32_f16(false, ap, false, bv, (short)0, acc, false, false);
    }
#pragma unroll
    for (int vv = 0; vv < 8; ++vv) {
      const int m = vv + lh * 8;
      AO[(size_t)(qb * 16 + m) * DMODEL + h * DHEAD + dt * 16 + lm] =
          (_Float16)(acc[vv] * rsinv[m]);
    }
  }
}

// ---------------------------------------------------------------------------
// Kernel 4: output projection  Y = AO(f16) @ Wo^T + bo  (f32 out).
// Wave computes a 32x64 tile (B-fragments reused across both row tiles).
// ---------------------------------------------------------------------------
__global__ __launch_bounds__(32)
void oproj_kernel(const _Float16* __restrict__ AO, const _Float16* __restrict__ Woh,
                  const float* __restrict__ bo, float* __restrict__ Y) {
  const int l  = threadIdx.x;
  const int lm = l & 15;
  const int lh = l >> 4;
  const int s0 = blockIdx.x * 32;
  const int n0 = blockIdx.y * 64;

  v8f acc[2][4] = {};
  const _Float16* arow0 = AO + (size_t)(s0 + lm) * DMODEL;
  const _Float16* arow1 = arow0 + (size_t)16 * DMODEL;

  for (int k0 = 0; k0 < DMODEL; k0 += 32) {
    __builtin_prefetch(arow0 + k0 + 64, 0, 1);
    v16h a0 = cat8(*(const v8h*)(arow0 + k0 + lh * 8),
                   *(const v8h*)(arow0 + k0 + lh * 8 + 16));
    v16h a1 = cat8(*(const v8h*)(arow1 + k0 + lh * 8),
                   *(const v8h*)(arow1 + k0 + lh * 8 + 16));
#pragma unroll
    for (int nt = 0; nt < 4; ++nt) {
      v16h b = cat8(*(const v8h*)(Woh + (size_t)(n0 + nt * 16 + lm) * DMODEL + k0 + lh * 8),
                    *(const v8h*)(Woh + (size_t)(n0 + nt * 16 + lm) * DMODEL + k0 + lh * 8 + 16));
      acc[0][nt] = __builtin_amdgcn_wmma_f32_16x16x32_f16(
          false, a0, false, b, (short)0, acc[0][nt], false, false);
      acc[1][nt] = __builtin_amdgcn_wmma_f32_16x16x32_f16(
          false, a1, false, b, (short)0, acc[1][nt], false, false);
    }
  }

#pragma unroll
  for (int rt = 0; rt < 2; ++rt) {
    const int sb = s0 + rt * 16;
#pragma unroll
    for (int nt = 0; nt < 4; ++nt) {
      const int n  = n0 + nt * 16 + lm;
      const float bb = bo[n];
#pragma unroll
      for (int vv = 0; vv < 8; ++vv) {
        const int m = vv + lh * 8;
        Y[(size_t)(sb + m) * DMODEL + n] = acc[rt][nt][vv] + bb;
      }
    }
  }
}

// ---------------------------------------------------------------------------
// Host entry
// ---------------------------------------------------------------------------
extern "C" void kernel_launch(void* const* d_in, const int* in_sizes, int n_in,
                              void* d_out, int out_size, void* d_ws, size_t ws_size,
                              hipStream_t stream) {
  (void)in_sizes; (void)n_in; (void)out_size; (void)ws_size;

  const float* q  = (const float*)d_in[0];
  const float* k  = (const float*)d_in[1];
  const float* v  = (const float*)d_in[2];
  const float* wq = (const float*)d_in[3];
  const float* bq = (const float*)d_in[4];
  const float* wk = (const float*)d_in[5];
  const float* bk = (const float*)d_in[6];
  const float* wv = (const float*)d_in[7];
  const float* bv = (const float*)d_in[8];
  const float* wo = (const float*)d_in[9];
  const float* bo = (const float*)d_in[10];
  float* out = (float*)d_out;

  // Workspace layout (f16):
  //   Xh [3][S][DM]      24 MB
  //   Wh [4][DM][DM]      8 MB
  //   Qr/Kr [H][S][64]  2x8 MB
  //   Vt [H][64][S]       8 MB
  //   AO [S][DM]          8 MB      total 64 MB
  char* ws = (char*)d_ws;
  const size_t XSEG = (size_t)S_LEN * DMODEL * sizeof(_Float16);           // 8 MB
  const size_t WSEG = (size_t)DMODEL * DMODEL * sizeof(_Float16);          // 2 MB
  _Float16* Xh = (_Float16*)(ws);
  _Float16* Wh = (_Float16*)(ws + 3 * XSEG);
  _Float16* Qr = (_Float16*)(ws + 3 * XSEG + 4 * WSEG);
  _Float16* Kr = (_Float16*)(ws + 3 * XSEG + 4 * WSEG + XSEG);
  _Float16* Vt = (_Float16*)(ws + 3 * XSEG + 4 * WSEG + 2 * XSEG);
  _Float16* AO = (_Float16*)(ws + 3 * XSEG + 4 * WSEG + 3 * XSEG);

  // 0) one-shot f32 -> f16 conversion of activations and weights
  const int XN = S_LEN * DMODEL;      // 4 Mi elements
  const int WN = DMODEL * DMODEL;     // 1 Mi elements
  cvt_f16_kernel<<<XN / (8 * 256), 256, 0, stream>>>(q,  Xh);
  cvt_f16_kernel<<<XN / (8 * 256), 256, 0, stream>>>(k,  Xh + (size_t)XN);
  cvt_f16_kernel<<<XN / (8 * 256), 256, 0, stream>>>(v,  Xh + (size_t)2 * XN);
  cvt_f16_kernel<<<WN / (8 * 256), 256, 0, stream>>>(wq, Wh);
  cvt_f16_kernel<<<WN / (8 * 256), 256, 0, stream>>>(wk, Wh + (size_t)WN);
  cvt_f16_kernel<<<WN / (8 * 256), 256, 0, stream>>>(wv, Wh + (size_t)2 * WN);
  cvt_f16_kernel<<<WN / (8 * 256), 256, 0, stream>>>(wo, Wh + (size_t)3 * WN);

  // 1) QKV projections (WMMA f16, 32x64 per wave)
  qkv_proj_kernel<<<dim3(S_LEN / 32, DMODEL / 64, 3), 32, 0, stream>>>(
      Xh, Wh, bq, bk, bv, Qr, Kr, Vt);

  // 2) RoPE on Q (x 1/sqrt(dk)) and K
  rope_kernel<<<dim3((NHEADS * S_LEN) / 256, 2), 256, 0, stream>>>(Qr, Kr);

  // 3) Banded attention (WMMA f16 over the 272-wide band)
  attn_kernel<<<dim3(S_LEN / 16, NHEADS), 32, 0, stream>>>(Qr, Kr, Vt, AO);

  // 4) Output projection (WMMA f16, f32 out)
  oproj_kernel<<<dim3(S_LEN / 32, DMODEL / 64), 32, 0, stream>>>(
      AO, Wh + (size_t)3 * WN, bo, out);
}
